// PairwiseCosineSimilarity_40200893890720
// MI455X (gfx1250) — compile-verified, matrix operand order checked by reading, not run on hardware
//
#include <hip/hip_runtime.h>
#include <hip/hip_bf16.h>
#include <math.h>

// ---------------------------------------------------------------------------
// Pairwise cosine similarity:  C[i,j] = <xn_i, yn_j>,  xn/yn = l2-normalized
// rows of x[8192,512], y[8192,512].
//   (1) wave-per-row normalize + bf16 hi/lo split into workspace
//   (2) tiled WMMA GEMM  C = Xn * Yn^T  (v_wmma_f32_16x16x32_bf16, bf16x3
//       split: hi*hi + hi*lo + lo*hi for near-f32 accuracy)
//   (3) K-loop staging via Tensor Data Mover (tensor_load_to_lds), LDS
//       double buffer with compile-time buffer indices; TDM fetch of step
//       s+1 overlaps the 24 WMMAs of step s.  C streams out with NT stores
//       so the 32MB of bf16 operands stay resident in the 192MB L2.
// ---------------------------------------------------------------------------

typedef __attribute__((ext_vector_type(16))) __bf16       v16bf;
typedef __attribute__((ext_vector_type(8)))  float        v8f;
typedef __attribute__((ext_vector_type(4)))  float        f32x4;
typedef __attribute__((ext_vector_type(4)))  unsigned int u32x4;
typedef __attribute__((ext_vector_type(8)))  int          i32x8;
typedef __attribute__((ext_vector_type(4)))  int          i32x4;

#define ROW_LEN 512          // L (fixed by reference)
#define BM      128          // workgroup tile M
#define BN      128          // workgroup tile N
#define BK      32           // K per WMMA step (bf16)
#define LSTR    40           // padded LDS row stride in bf16 (80B -> conflict-free b128 reads)
#define KSTEPS  (ROW_LEN / BK)   // 16 (even; loop processes step pairs)

#if defined(__has_builtin)
# if __has_builtin(__builtin_amdgcn_tensor_load_to_lds)
#  define USE_TDM 1
# endif
#endif
#ifndef USE_TDM
# define USE_TDM 0
#endif

// ---------------------------------------------------------------------------
// Kernel 1: normalize rows, split into bf16 hi + lo (wave per row).
// ---------------------------------------------------------------------------
union B4Pack { __bf16 h[4]; unsigned long long u; };

__global__ __launch_bounds__(256) void normalize_split_kernel(
    const float* __restrict__ src,
    unsigned short* __restrict__ hi_out,
    unsigned short* __restrict__ lo_out,
    int nrows)
{
    const int lane = threadIdx.x & 31;
    const int row  = blockIdx.x * (blockDim.x >> 5) + (threadIdx.x >> 5);
    if (row >= nrows) return;

    const f32x4* rp = (const f32x4*)(src + (size_t)row * ROW_LEN);
    f32x4 v[4];
    float s = 0.0f;
#pragma unroll
    for (int i = 0; i < 4; ++i) {
        v[i] = __builtin_nontemporal_load(rp + lane + 32 * i);  // one-shot read
        s += v[i].x * v[i].x + v[i].y * v[i].y + v[i].z * v[i].z + v[i].w * v[i].w;
    }
#pragma unroll
    for (int off = 16; off > 0; off >>= 1)
        s += __shfl_xor(s, off, 32);

    const float scale = 1.0f / fmaxf(sqrtf(s), 1e-8f);

    __bf16* hi = (__bf16*)hi_out;
    __bf16* lo = (__bf16*)lo_out;
#pragma unroll
    for (int i = 0; i < 4; ++i) {
        B4Pack ph, pl;
#pragma unroll
        for (int c = 0; c < 4; ++c) {
            float f  = v[i][c] * scale;
            __bf16 fh = (__bf16)f;                 // RNE convert
            float  fl = f - (float)fh;             // residual
            ph.h[c] = fh;
            pl.h[c] = (__bf16)fl;
        }
        const size_t idx = (size_t)row * ROW_LEN + (size_t)(lane + 32 * i) * 4;
        *(unsigned long long*)(hi + idx) = ph.u;   // regular TH: GEMM re-reads these
        *(unsigned long long*)(lo + idx) = pl.u;
    }
}

// ---------------------------------------------------------------------------
// TDM: DMA one 128-row x 32-col bf16 tile (row pitch ROW_LEN) into LDS with
// hardware row padding 64B data + 16B pad  ->  LSTR(40 bf16) stride.
// D# layout per CDNA5 ISA.
// ---------------------------------------------------------------------------
__device__ __forceinline__ void tdm_tile_load(const void* gptr, unsigned lds_off,
                                              unsigned tensor_rows)
{
#if USE_TDM
    const unsigned long long ga = (unsigned long long)gptr;

    u32x4 g0;
    g0[0] = 1u;                                          // count=1, user descriptor
    g0[1] = lds_off;                                     // lds_addr (bytes)
    g0[2] = (unsigned)(ga & 0xFFFFFFFFu);                // global_addr[31:0]
    g0[3] = (unsigned)((ga >> 32) & 0x01FFFFFFu)         // global_addr[56:32]
          | (2u << 30);                                  // type = 2 (image)

    i32x8 g1;
    g1[0] = (1 << 16)                                    // data_size = 1 -> 2 bytes
          | (1 << 20)                                    // pad_enable
          | (3 << 22)                                    // pad_interval = 16 DWORDs (64B)
          | (3 << 25);                                   // pad_amount   = 4 DWORDs (16B)
    g1[1] = (int)((ROW_LEN & 0xFFFFu) << 16);            // tensor_dim0[15:0] @ bits 63:48
    g1[2] = (int)((ROW_LEN >> 16)                        // tensor_dim0[31:16]
          | ((tensor_rows & 0xFFFFu) << 16));            // tensor_dim1[15:0]
    g1[3] = (int)((tensor_rows >> 16)                    // tensor_dim1[31:16]
          | (BK << 16));                                 // tile_dim0 = 32 @ bits 127:112
    g1[4] = (int)BM;                                     // tile_dim1 = 128, tile_dim2 = 0
    g1[5] = (int)ROW_LEN;                                // tensor_dim0_stride[31:0]
    g1[6] = 0;                                           // stride0 hi16 | stride1 lo16
    g1[7] = 0;                                           // stride1 hi32

    i32x4 z4 = {0, 0, 0, 0};                             // groups 2/3 unused (2D)
# if __clang_major__ >= 23
    i32x8 z8 = {0, 0, 0, 0, 0, 0, 0, 0};
    __builtin_amdgcn_tensor_load_to_lds(g0, g1, z4, z4, z8, 0);
# else
    __builtin_amdgcn_tensor_load_to_lds(g0, g1, z4, z4, 0);
# endif
#else
    (void)gptr; (void)lds_off; (void)tensor_rows;
#endif
}

// ---------------------------------------------------------------------------
// Fragment loader: ISA 16-bit A-matrix 16x32 layout.
//   lanes 0-15 : elems 0..7 = K{0..7},  elems 8..15 = K{16..23}
//   lanes 16-31: elems 0..7 = K{8..15}, elems 8..15 = K{24..31}
// ---------------------------------------------------------------------------
union FragCast { v16bf v; u32x4 q[2]; };

__device__ __forceinline__ v16bf load_frag(const __bf16* lds, int r, int klo)
{
    FragCast f;
    f.q[0] = *(const u32x4*)(lds + r * LSTR + klo);
    f.q[1] = *(const u32x4*)(lds + r * LSTR + klo + 16);
    return f.v;
}

__device__ __forceinline__ v8f bf16_wmma(v16bf a, v16bf b, v8f c)
{
    return __builtin_amdgcn_wmma_f32_16x16x32_bf16(
        false, a, false, b, (short)0, c, false, false);
}

// One K-step of compute on a staged 4-tile LDS buffer (base is a constant
// address after inlining, so all DS offsets fold to immediates).
__device__ __forceinline__ void compute_step(const __bf16* base,
                                             int wm, int wn, int mrow, int klo,
                                             v8f (&acc)[4][2])
{
    const __bf16* sAh = base;
    const __bf16* sAl = base + 1 * BM * LSTR;
    const __bf16* sBh = base + 2 * BM * LSTR;
    const __bf16* sBl = base + 3 * BM * LSTR;

    v16bf ah[4], al[4], bh[2], bl[2];
#pragma unroll
    for (int m = 0; m < 4; ++m) {
        const int r = wm + m * 16 + mrow;
        ah[m] = load_frag(sAh, r, klo);
        al[m] = load_frag(sAl, r, klo);
    }
#pragma unroll
    for (int n = 0; n < 2; ++n) {
        const int r = wn + n * 16 + mrow;
        bh[n] = load_frag(sBh, r, klo);
        bl[n] = load_frag(sBl, r, klo);
    }

#pragma unroll
    for (int m = 0; m < 4; ++m)
#pragma unroll
        for (int n = 0; n < 2; ++n) {
            acc[m][n] = bf16_wmma(ah[m], bh[n], acc[m][n]);  // hi*hi
            acc[m][n] = bf16_wmma(ah[m], bl[n], acc[m][n]);  // hi*lo
            acc[m][n] = bf16_wmma(al[m], bh[n], acc[m][n]);  // lo*hi
        }
}

// ---------------------------------------------------------------------------
// Kernel 2: C = Xn * Yn^T, bf16x3 split accumulation.
// 256 threads = 8 waves; tile 128x128; wave tile 64x32 (4x2 WMMA tiles).
// Double-buffered LDS (2 x 4 tiles x 10240B = 80KB) + async TDM prefetch.
// ---------------------------------------------------------------------------
__global__ __launch_bounds__(256) void cosine_wmma_kernel(
    const unsigned short* __restrict__ Ahi_,
    const unsigned short* __restrict__ Alo_,
    const unsigned short* __restrict__ Bhi_,
    const unsigned short* __restrict__ Blo_,
    float* __restrict__ C,
    int nrowsA, int nrowsB, int ldc)
{
    const __bf16* Ahi = (const __bf16*)Ahi_;
    const __bf16* Alo = (const __bf16*)Alo_;
    const __bf16* Bhi = (const __bf16*)Bhi_;
    const __bf16* Blo = (const __bf16*)Blo_;

    __shared__ __attribute__((aligned(16))) __bf16 smem[2][4][BM * LSTR];

    const int tid  = threadIdx.x;
    const int lane = tid & 31;
    const int wave = tid >> 5;
    const int wm   = (wave >> 2) * 64;   // wave M offset: 0 / 64
    const int wn   = (wave & 3) * 32;    // wave N offset: 0/32/64/96

    const int blockM = blockIdx.y * BM;
    const int blockN = blockIdx.x * BN;

    const int klo  = (lane < 16) ? 0 : 8;
    const int mrow = lane & 15;

    v8f acc[4][2] = {};

#if USE_TDM
    // ---- TDM double-buffered pipeline, explicit step pairs ----------------
    auto issue_tiles = [&](int buf, int k0) {
        const size_t a0 = (size_t)blockM * ROW_LEN + k0;
        const size_t b0 = (size_t)blockN * ROW_LEN + k0;
        tdm_tile_load(Ahi + a0, (unsigned)(unsigned long long)&smem[buf][0][0], nrowsA);
        tdm_tile_load(Alo + a0, (unsigned)(unsigned long long)&smem[buf][1][0], nrowsA);
        tdm_tile_load(Bhi + b0, (unsigned)(unsigned long long)&smem[buf][2][0], nrowsB);
        tdm_tile_load(Blo + b0, (unsigned)(unsigned long long)&smem[buf][3][0], nrowsB);
    };

    if (wave == 0) issue_tiles(0, 0);

#pragma unroll 1
    for (int s = 0; s < KSTEPS; s += 2) {
        // ---- step s: compute buffer 0, prefetch into buffer 1 ----
        if (wave == 0)
            __builtin_amdgcn_s_wait_tensorcnt(0);     // buf0 resident
        __syncthreads();                              // readiness + buf1 reads done
        if (wave == 0 && (s + 1) < KSTEPS)
            issue_tiles(1, (s + 1) * BK);
        compute_step(&smem[0][0][0], wm, wn, mrow, klo, acc);

        // ---- step s+1: compute buffer 1, prefetch into buffer 0 ----
        if (wave == 0)
            __builtin_amdgcn_s_wait_tensorcnt(0);     // buf1 resident
        __syncthreads();                              // readiness + buf0 reads done
        if (wave == 0 && (s + 2) < KSTEPS)
            issue_tiles(0, (s + 2) * BK);
        compute_step(&smem[1][0][0], wm, wn, mrow, klo, acc);
    }
#else
    // ---- Fallback: synchronous VGPR-bounce staging (single buffer) --------
    for (int k0 = 0; k0 < ROW_LEN; k0 += BK) {
        __syncthreads();
        for (int u = tid; u < (BM * BK) / 8; u += 256) {
            const int r   = u >> 2;
            const int seg = (u & 3) << 3;
            const size_t ga = (size_t)(blockM + r) * ROW_LEN + k0 + seg;
            const size_t gb = (size_t)(blockN + r) * ROW_LEN + k0 + seg;
            *(u32x4*)(&smem[0][0][0] + r * LSTR + seg) = *(const u32x4*)(Ahi + ga);
            *(u32x4*)(&smem[0][1][0] + r * LSTR + seg) = *(const u32x4*)(Alo + ga);
            *(u32x4*)(&smem[0][2][0] + r * LSTR + seg) = *(const u32x4*)(Bhi + gb);
            *(u32x4*)(&smem[0][3][0] + r * LSTR + seg) = *(const u32x4*)(Blo + gb);
        }
        __syncthreads();
        compute_step(&smem[0][0][0], wm, wn, mrow, klo, acc);
    }
#endif

    // Epilogue: C/D 16x16 f32 layout — VGPR r: M=r (lanes 0-15), M=r+8 (16-31).
    // NT stores: C is a 256MB one-shot stream; keep operands hot in L2 instead.
    const int crow = (lane >> 4) << 3;
    const int ccol = lane & 15;
#pragma unroll
    for (int m = 0; m < 4; ++m)
#pragma unroll
        for (int n = 0; n < 2; ++n)
#pragma unroll
            for (int r = 0; r < 8; ++r) {
                const size_t gr = (size_t)(blockM + wm + m * 16 + crow + r);
                const size_t gc = (size_t)(blockN + wn + n * 16 + ccol);
                __builtin_nontemporal_store(acc[m][n][r], &C[gr * (size_t)ldc + gc]);
            }
}

// ---------------------------------------------------------------------------
// Launcher
// ---------------------------------------------------------------------------
extern "C" void kernel_launch(void* const* d_in, const int* in_sizes, int n_in,
                              void* d_out, int out_size, void* d_ws, size_t ws_size,
                              hipStream_t stream)
{
    const float* x = (const float*)d_in[0];
    const float* y = (const float*)d_in[1];
    const int Nx = in_sizes[0] / ROW_LEN;   // 8192
    const int Ny = in_sizes[1] / ROW_LEN;   // 8192

    // Workspace: xhi | xlo | yhi | ylo  (bf16 each, 32 MB total)
    unsigned short* xhi = (unsigned short*)d_ws;
    unsigned short* xlo = xhi + (size_t)Nx * ROW_LEN;
    unsigned short* yhi = xlo + (size_t)Nx * ROW_LEN;
    unsigned short* ylo = yhi + (size_t)Ny * ROW_LEN;

    const int rows_per_block = 256 / 32;
    normalize_split_kernel<<<(Nx + rows_per_block - 1) / rows_per_block, 256, 0, stream>>>(
        x, xhi, xlo, Nx);
    normalize_split_kernel<<<(Ny + rows_per_block - 1) / rows_per_block, 256, 0, stream>>>(
        y, yhi, ylo, Ny);

    dim3 grid(Ny / BN, Nx / BM);   // (64, 64)
    cosine_wmma_kernel<<<grid, 256, 0, stream>>>(xhi, xlo, yhi, ylo,
                                                 (float*)d_out, Nx, Ny, Ny);
}
